// Attention2Neighbors_64364379898307
// MI455X (gfx1250) — matrix operand order repaired and checked by reading, not compiled, not run on hardware
//
#include <hip/hip_runtime.h>
#include <stdint.h>

// ---------------------------------------------------------------------------
// MHSA for MI455X (gfx1250): bf16 WMMA everywhere, fp32 accumulate.
// Shapes fixed by the reference: B=4, G=2048, C=1024, H=16, hd=64, scale=1/8.
// Data path: fp32 -> bf16 once; all 4 matmuls via v_wmma_f32_16x16x32_bf16;
// LDS staging via global_load_async_to_lds_b128 (ASYNCcnt).
// ---------------------------------------------------------------------------

typedef __attribute__((ext_vector_type(16))) __bf16        bf16x16;
typedef __attribute__((ext_vector_type(8)))  float         f32x8;
typedef __attribute__((ext_vector_type(8)))  unsigned int  u32x8;
typedef int vint4 __attribute__((vector_size(16)));   // matches builtin's V4i

#define AS1 __attribute__((address_space(1)))
#define AS3 __attribute__((address_space(3)))

#if defined(__AMDGCN__) && __has_builtin(__builtin_amdgcn_global_load_async_to_lds_b128)
#define USE_ASYNC_LDS 1
#else
#define USE_ASYNC_LDS 0
#endif

static __device__ __forceinline__ unsigned short f2bf(float f) {
  unsigned int u = __builtin_bit_cast(unsigned int, f);
  unsigned int r = u + 0x7FFFu + ((u >> 16) & 1u);   // round-to-nearest-even
  return (unsigned short)(r >> 16);
}

static __device__ __forceinline__ f32x8 wmma_bf16(u32x8 a, u32x8 b, f32x8 c) {
  return __builtin_amdgcn_wmma_f32_16x16x32_bf16(
      false, __builtin_bit_cast(bf16x16, a),
      false, __builtin_bit_cast(bf16x16, b),
      (short)0, c, false, false);
}

// A-matrix (16x32, 16-bit) K offset for vgpr v, this lane (ISA 7.12.2)
static __device__ __forceinline__ int ka_off(int v, int lane) {
  return ((v >> 2) << 4) | (((lane >> 4) & 1) << 3) | ((v & 3) << 1);
}
// B-matrix (32x16, 16-bit), column n = lane&15, K order matches A (ka_off)
static __device__ __forceinline__ int kb_off(int v, int lane) {
  return ((v >> 2) << 4) | (((lane >> 4) & 1) << 3) | ((v & 3) << 1);
}

// ---- staging: 16 bf16 (32B) global -> LDS -----------------------------------
static __device__ __forceinline__ void stage16(const unsigned short* g,
                                               unsigned short* l) {
#if USE_ASYNC_LDS
  __builtin_amdgcn_global_load_async_to_lds_b128(
      (AS1 vint4*)g,       (AS3 vint4*)l,       0, 0);
  __builtin_amdgcn_global_load_async_to_lds_b128(
      (AS1 vint4*)(g + 8), (AS3 vint4*)(l + 8), 0, 0);
#else
  const uint4* p = (const uint4*)g;
  uint4 d0 = p[0], d1 = p[1];
  unsigned int* s = (unsigned int*)l;
  s[0]=d0.x; s[1]=d0.y; s[2]=d0.z; s[3]=d0.w;
  s[4]=d1.x; s[5]=d1.y; s[6]=d1.z; s[7]=d1.w;
#endif
}

static __device__ __forceinline__ void wait_async() {
#if USE_ASYNC_LDS
 #if __has_builtin(__builtin_amdgcn_s_wait_asynccnt)
  __builtin_amdgcn_s_wait_asynccnt(0);
 #else
  asm volatile("s_wait_asynccnt 0x0" ::: "memory");
 #endif
#endif
}

// ---------------------------------------------------------------------------
// Kernel 1: fp32 -> bf16 convert
// ---------------------------------------------------------------------------
__global__ __launch_bounds__(256) void cvt_f32_bf16(
    const float* __restrict__ src, unsigned short* __restrict__ dst, int n) {
  int i = blockIdx.x * 256 + threadIdx.x;
  int stride = gridDim.x * 256;
  for (; i < n; i += stride) dst[i] = f2bf(src[i]);
}

// ---------------------------------------------------------------------------
// Kernel 2/4: C = A (MxK) * W^T (W is NxK row-major), bf16 in.
// 256 thr (8 waves), 128x128 tile, K-step 32, double-buffered async LDS.
// Wave (wm 0..1, wn 0..3) owns 64x32 = 4x2 WMMA tiles.
// grid.z strides W and Cb for the fused q/k/v launch.
// ---------------------------------------------------------------------------
#define LSTR 34   // LDS row pitch in bf16 (17 dwords -> conflict-free b32 reads)

template <bool F32OUT>
__global__ __launch_bounds__(256) void gemm_xWt(
    const unsigned short* __restrict__ A,
    const unsigned short* __restrict__ W0,
    unsigned short* __restrict__ Cb0,      // bf16 out (F32OUT==false)
    float* __restrict__ Cf,                // f32 out  (F32OUT==true)
    const float* __restrict__ bias,        // with F32OUT
    int M, int N, int K)
{
  __shared__ unsigned short As[2][128 * LSTR];
  __shared__ unsigned short Ws[2][128 * LSTR];

  const unsigned short* W  = W0 + (size_t)blockIdx.z * (size_t)N * K;
  unsigned short*       Cb = Cb0 + (size_t)blockIdx.z * (size_t)M * N;

  const int tid  = threadIdx.x;
  const int lane = tid & 31;
  const int wave = tid >> 5;
  const int wm   = wave >> 2;            // 0..1
  const int wn   = wave & 3;             // 0..3
  const int m0   = blockIdx.x * 128;
  const int n0   = blockIdx.y * 128;
  const int lrow = tid >> 1;             // staging row 0..127
  const int soff = lrow * LSTR + (tid & 1) * 16;

  f32x8 acc[4][2] = {};

  const unsigned short* gA = A + (size_t)(m0 + lrow) * K + (tid & 1) * 16;
  const unsigned short* gW = W + (size_t)(n0 + lrow) * K + (tid & 1) * 16;

  // prologue: stage first K-slab into buffer 0
  stage16(gA, &As[0][soff]);
  stage16(gW, &Ws[0][soff]);

  int buf = 0;
  for (int k0 = 0; k0 < K; k0 += 32) {
    wait_async();                 // own async fills done
    __syncthreads();              // whole tile visible; prior buf consumed
    if (k0 + 32 < K) {            // overlap next slab's DMA with compute
      stage16(gA + k0 + 32, &As[buf ^ 1][soff]);
      stage16(gW + k0 + 32, &Ws[buf ^ 1][soff]);
      __builtin_prefetch(gA + k0 + 64, 0, 1);   // global_prefetch_b8
      __builtin_prefetch(gW + k0 + 64, 0, 1);
    }

    u32x8 af[4];
#pragma unroll
    for (int mt = 0; mt < 4; ++mt) {
      const unsigned short* p = &As[buf][(wm * 64 + mt * 16 + (lane & 15)) * LSTR];
#pragma unroll
      for (int v = 0; v < 8; ++v)
        af[mt][v] = *(const unsigned int*)&p[ka_off(v, lane)];
    }
    u32x8 wf[2];
#pragma unroll
    for (int nt = 0; nt < 2; ++nt) {
      const unsigned short* p = &Ws[buf][(wn * 32 + nt * 16 + (lane & 15)) * LSTR];
#pragma unroll
      for (int v = 0; v < 8; ++v)
        wf[nt][v] = *(const unsigned int*)&p[kb_off(v, lane)];
    }
#pragma unroll
    for (int mt = 0; mt < 4; ++mt)
#pragma unroll
      for (int nt = 0; nt < 2; ++nt)
        acc[mt][nt] = wmma_bf16(af[mt], wf[nt], acc[mt][nt]);
    buf ^= 1;
  }

  // epilogue: C/D layout -> row = r + 8*(lane>>4), col = lane&15
  const int lr = (lane >> 4) * 8;
  const int lc = lane & 15;
#pragma unroll
  for (int mt = 0; mt < 4; ++mt)
#pragma unroll
    for (int nt = 0; nt < 2; ++nt)
#pragma unroll
      for (int r = 0; r < 8; ++r) {
        size_t row = (size_t)(m0 + wm * 64 + mt * 16 + lr + r);
        int    col = n0 + wn * 32 + nt * 16 + lc;
        float  val = acc[mt][nt][r];
        if constexpr (F32OUT) Cf[row * N + col] = val + bias[col];
        else                  Cb[row * N + col] = f2bf(val);
      }
}

// ---------------------------------------------------------------------------
// Kernel 3: flash attention, one WG per (64-query tile, b*H+h).
// 128 thr (4 waves); each wave owns 16 query rows x hd=64.
// K-tile async-staged row-major; V-tile transposed in LDS so O=PV B-fragments
// are contiguous pairs; P flips C-layout -> A-layout through per-wave LDS.
// ---------------------------------------------------------------------------
__global__ __launch_bounds__(128) void flash_attn(
    const unsigned short* __restrict__ Q,    // (B*G, C) bf16
    const unsigned short* __restrict__ Kk,
    const unsigned short* __restrict__ V,
    unsigned short* __restrict__ O,          // (B*G, C) bf16
    int G, int C, int H)
{
  __shared__ unsigned short Ks [64 * LSTR];       // K rows x hd
  __shared__ unsigned short Vts[64 * LSTR];       // V transposed: d x key
  __shared__ unsigned short Ps [4][16 * LSTR];    // per-wave P tile

  const int tid  = threadIdx.x;
  const int lane = tid & 31;
  const int wave = tid >> 5;
  const int b    = blockIdx.y / H;
  const int h    = blockIdx.y % H;
  const size_t rowbase = (size_t)b * G;
  const int hoff = h * 64;
  const int lr   = (lane >> 4) * 8;
  const int lc   = lane & 15;

  // Q fragments (A-matrix), 2 K-steps over hd=64
  u32x8 qa[2];
  {
    const unsigned short* qp =
        Q + (rowbase + blockIdx.x * 64 + wave * 16 + lc) * C + hoff;
#pragma unroll
    for (int ds = 0; ds < 2; ++ds)
#pragma unroll
      for (int v = 0; v < 8; ++v)
        qa[ds][v] = *(const unsigned int*)&qp[ds * 32 + ka_off(v, lane)];
  }

  f32x8 oacc[4] = {};
  float rm[8], rs[8];
#pragma unroll
  for (int r = 0; r < 8; ++r) { rm[r] = -1e30f; rs[r] = 0.0f; }

  const int srow  = tid >> 1;     // staging row 0..63
  const int shalf = tid & 1;      // 32-elem half

  for (int kb = 0; kb < G / 64; ++kb) {
    // ---- V rows into regs (overlaps prior compute tail) ----
    const unsigned short* kp =
        Kk + (rowbase + (size_t)kb * 64 + srow) * C + hoff + shalf * 32;
    const unsigned short* vp =
        V  + (rowbase + (size_t)kb * 64 + srow) * C + hoff + shalf * 32;
    const uint4* vp4 = (const uint4*)vp;
    uint4 vq[4] = { vp4[0], vp4[1], vp4[2], vp4[3] };
    unsigned int vw[16] = { vq[0].x,vq[0].y,vq[0].z,vq[0].w,
                            vq[1].x,vq[1].y,vq[1].z,vq[1].w,
                            vq[2].x,vq[2].y,vq[2].z,vq[2].w,
                            vq[3].x,vq[3].y,vq[3].z,vq[3].w };
    __syncthreads();   // all prior-iteration LDS reads complete

    // ---- K tile: async DMA to LDS; V tile: transposed scalar stores ----
    unsigned short* sk = &Ks[srow * LSTR + shalf * 32];
    stage16(kp,      sk);
    stage16(kp + 16, sk + 16);
#pragma unroll
    for (int j = 0; j < 16; ++j) {
      int d = shalf * 32 + 2 * j;
      Vts[(d    ) * LSTR + srow] = (unsigned short)(vw[j] & 0xFFFFu);
      Vts[(d + 1) * LSTR + srow] = (unsigned short)(vw[j] >> 16);
    }
    wait_async();
    __syncthreads();

    // ---- S = (Q K^T) * scale ----
    f32x8 sacc[4] = {};
#pragma unroll
    for (int nt = 0; nt < 4; ++nt) {
      const unsigned short* p = &Ks[(nt * 16 + lc) * LSTR];
#pragma unroll
      for (int ds = 0; ds < 2; ++ds) {
        u32x8 bf;
#pragma unroll
        for (int v = 0; v < 8; ++v)
          bf[v] = *(const unsigned int*)&p[ds * 32 + kb_off(v, lane)];
        sacc[nt] = wmma_bf16(qa[ds], bf, sacc[nt]);
      }
    }
#pragma unroll
    for (int nt = 0; nt < 4; ++nt)
#pragma unroll
      for (int r = 0; r < 8; ++r) sacc[nt][r] *= 0.125f;   // hd^-0.5

    // ---- online softmax (rows live in 16-lane halves) ----
    float mx[8];
#pragma unroll
    for (int r = 0; r < 8; ++r) {
      mx[r] = fmaxf(fmaxf(sacc[0][r], sacc[1][r]),
                    fmaxf(sacc[2][r], sacc[3][r]));
#pragma unroll
      for (int m = 1; m < 16; m <<= 1)
        mx[r] = fmaxf(mx[r], __shfl_xor(mx[r], m, 32));
    }
#pragma unroll
    for (int r = 0; r < 8; ++r) {
      float newm = fmaxf(rm[r], mx[r]);
      float corr = __expf(rm[r] - newm);
      rs[r] *= corr;
#pragma unroll
      for (int nt = 0; nt < 4; ++nt) oacc[nt][r] *= corr;
      rm[r] = newm;
    }
    float psum[8];
#pragma unroll
    for (int r = 0; r < 8; ++r) psum[r] = 0.0f;
#pragma unroll
    for (int nt = 0; nt < 4; ++nt)
#pragma unroll
      for (int r = 0; r < 8; ++r) {
        float p = __expf(sacc[nt][r] - rm[r]);
        sacc[nt][r] = p;
        psum[r] += p;
      }
#pragma unroll
    for (int r = 0; r < 8; ++r) {
#pragma unroll
      for (int m = 1; m < 16; m <<= 1) psum[r] += __shfl_xor(psum[r], m, 32);
      rs[r] += psum[r];
    }

    // ---- P: C-layout -> LDS -> A-layout ----
#pragma unroll
    for (int nt = 0; nt < 4; ++nt)
#pragma unroll
      for (int r = 0; r < 8; ++r)
        Ps[wave][(lr + r) * LSTR + nt * 16 + lc] = f2bf(sacc[nt][r]);
    __syncthreads();

    // ---- O += P * V ----
#pragma unroll
    for (int ds = 0; ds < 2; ++ds) {
      u32x8 pa;
      const unsigned short* pp = &Ps[wave][lc * LSTR];
#pragma unroll
      for (int v = 0; v < 8; ++v)
        pa[v] = *(const unsigned int*)&pp[ds * 32 + ka_off(v, lane)];
#pragma unroll
      for (int nt = 0; nt < 4; ++nt) {
        const unsigned short* vp2 = &Vts[(nt * 16 + lc) * LSTR];
        u32x8 bf;
#pragma unroll
        for (int v = 0; v < 8; ++v)
          bf[v] = *(const unsigned int*)&vp2[ds * 32 + kb_off(v, lane)];
        oacc[nt] = wmma_bf16(pa, bf, oacc[nt]);
      }
    }
    // next iteration's pre-store barrier protects Ks/Vts/Ps overwrite
  }

  // ---- normalize + store bf16 (merged-head layout h*64+d) ----
#pragma unroll
  for (int nt = 0; nt < 4; ++nt)
#pragma unroll
    for (int r = 0; r < 8; ++r) {
      size_t g = rowbase + blockIdx.x * 64 + wave * 16 + lr + r;
      int    c = hoff + nt * 16 + lc;
      O[g * C + c] = f2bf(oacc[nt][r] / rs[r]);
    }
}

// ---------------------------------------------------------------------------
// Host-side orchestration
// ---------------------------------------------------------------------------
extern "C" void kernel_launch(void* const* d_in, const int* in_sizes, int n_in,
                              void* d_out, int out_size, void* d_ws, size_t ws_size,
                              hipStream_t stream) {
  (void)in_sizes; (void)n_in; (void)out_size; (void)ws_size;
  const int B = 4, G = 2048, C = 1024, H = 16;
  const int M  = B * G;                 // 8192
  const size_t XN = (size_t)M * C;      // 8,388,608
  const size_t WN = (size_t)C * C;      // 1,048,576

  const float* x  = (const float*)d_in[0];
  const float* Wq = (const float*)d_in[1];
  const float* Wk = (const float*)d_in[2];
  const float* Wv = (const float*)d_in[3];
  const float* Wp = (const float*)d_in[4];
  const float* bp = (const float*)d_in[5];
  float* out = (float*)d_out;

  // workspace layout (bf16/ushort units):
  //   [xb | wq wk wv | wp | q k v]; o aliases xb (x dead after QKV GEMM)
  unsigned short* xb  = (unsigned short*)d_ws;
  unsigned short* wqb = xb  + XN;            // wk = wqb+WN, wv = wqb+2*WN
  unsigned short* wpb = wqb + 3 * WN;
  unsigned short* qb  = wpb + WN;            // k = qb+XN, v = qb+2*XN
  unsigned short* ob  = xb;                  // alias

  // 1) fp32 -> bf16
  cvt_f32_bf16<<<4096, 256, 0, stream>>>(x,  xb,           (int)XN);
  cvt_f32_bf16<<<1024, 256, 0, stream>>>(Wq, wqb,          (int)WN);
  cvt_f32_bf16<<<1024, 256, 0, stream>>>(Wk, wqb + WN,     (int)WN);
  cvt_f32_bf16<<<1024, 256, 0, stream>>>(Wv, wqb + 2 * WN, (int)WN);
  cvt_f32_bf16<<<1024, 256, 0, stream>>>(Wp, wpb,          (int)WN);

  // 2) fused q/k/v projection: z strides W and output
  gemm_xWt<false><<<dim3(M / 128, C / 128, 3), 256, 0, stream>>>(
      xb, wqb, qb, nullptr, nullptr, M, C, C);

  // 3) flash attention per (64-query tile, b*H+h)
  flash_attn<<<dim3(G / 64, B * H), 128, 0, stream>>>(
      qb, qb + XN, qb + 2 * XN, ob, G, C, H);

  // 4) output projection with bias, fp32 out
  gemm_xWt<true><<<dim3(M / 128, C / 128, 1), 256, 0, stream>>>(
      ob, wpb, nullptr, out, bp, M, C, C);
}
